// GCN_87711822119195
// MI455X (gfx1250) — matrix operand order repaired and checked by reading, not compiled, not run on hardware
//
#include <hip/hip_runtime.h>
#include <hip/hip_bf16.h>

typedef __attribute__((ext_vector_type(16))) _Float16 v16h;
typedef __attribute__((ext_vector_type(8)))  float    v8f;

#define F 64

// ---------------- degree / norm prep ----------------
__global__ __launch_bounds__(256) void k_init_deg(float* deg, int n) {
  int i = blockIdx.x * 256 + threadIdx.x;
  if (i < n) deg[i] = 1.0f;                       // self-loop contributes 1
}

__global__ __launch_bounds__(256) void k_count_deg(const int* __restrict__ dst,
                                                   float* deg, int e) {
  int i = blockIdx.x * 256 + threadIdx.x;
  if (i < e) unsafeAtomicAdd(&deg[dst[i]], 1.0f); // hw global_atomic_add_f32
}

__global__ __launch_bounds__(256) void k_dinv(float* d, int n) {
  int i = blockIdx.x * 256 + threadIdx.x;
  if (i < n) d[i] = rsqrtf(d[i]);                 // deg >= 1 always
}

__global__ __launch_bounds__(256) void k_norm(const int* __restrict__ src,
                                              const int* __restrict__ dst,
                                              const float* __restrict__ dinv,
                                              float* __restrict__ norm, int e) {
  int i = blockIdx.x * 256 + threadIdx.x;
  if (i < e) norm[i] = dinv[src[i]] * dinv[dst[i]];
}

// ---------------- W -> WMMA B-fragment pre-pack ------------------------------
// Fragment (ks,nt), lane: 16 halves stored contiguously so the GEMM can load
// each v16h with two b128 loads. Contiguous k (this file's order, matches the A load): lane = column
// nt*16+l16; lanes 0-15 hold K=kr..kr+15, lanes 16-31 hold K=kr+16..kr+31.
__global__ __launch_bounds__(256) void k_prep_w(const float* __restrict__ W,
                                                _Float16* __restrict__ Wh) {
  const int tid  = threadIdx.x;
  const int lane = tid & 31;
  const int nt   = (tid >> 5) & 3;
  const int ks   = tid >> 7;
  const int l16  = lane & 15;
  const int hi   = lane >> 4;
  const int col  = nt * 16 + l16;
  const int kr   = ks * 32 + hi * 16;
  _Float16* o = Wh + (size_t)((ks * 4 + nt) * 32 + lane) * 16;
#pragma unroll
  for (int i = 0; i < 16; ++i)
    o[i] = (_Float16)W[(kr + i) * F + col];
}

// ---------------- WMMA GEMM: T = (relu?)X @ W --------------------------------
// Branch-free: A row clamped to n-1 (tail rows duplicate, never consumed);
// T is padded to a multiple of 512 rows so all stores are in-bounds.
template <int RELU>
__global__ __launch_bounds__(256) void k_gemm64(const float* __restrict__ X,
                                                const _Float16* __restrict__ Wh,
                                                float* __restrict__ T, int n) {
  const int wave = threadIdx.x >> 5;
  const int lane = threadIdx.x & 31;
  const int l16  = lane & 15;
  const int hi   = lane >> 4;

  v16h bf[2][4];
#pragma unroll
  for (int ks = 0; ks < 2; ++ks)
#pragma unroll
    for (int nt = 0; nt < 4; ++nt)
      bf[ks][nt] = *(const v16h*)(Wh + (size_t)((ks * 4 + nt) * 32 + lane) * 16);

  const int baseRow = (blockIdx.x * 8 + wave) * 64;
#pragma unroll
  for (int t = 0; t < 4; ++t) {
    const int row0 = baseRow + t * 16;
    int arow = row0 + l16;
    arow = arow < n ? arow : n - 1;                 // clamp: loads always valid
    const float* xr = X + (size_t)arow * F;

    v8f acc[4] = {};
#pragma unroll
    for (int ks = 0; ks < 2; ++ks) {
      const int kb = ks * 32 + hi * 16;
      float4 q0 = *(const float4*)(xr + kb);
      float4 q1 = *(const float4*)(xr + kb + 4);
      float4 q2 = *(const float4*)(xr + kb + 8);
      float4 q3 = *(const float4*)(xr + kb + 12);
      if (RELU) {
        q0.x = fmaxf(q0.x, 0.f); q0.y = fmaxf(q0.y, 0.f);
        q0.z = fmaxf(q0.z, 0.f); q0.w = fmaxf(q0.w, 0.f);
        q1.x = fmaxf(q1.x, 0.f); q1.y = fmaxf(q1.y, 0.f);
        q1.z = fmaxf(q1.z, 0.f); q1.w = fmaxf(q1.w, 0.f);
        q2.x = fmaxf(q2.x, 0.f); q2.y = fmaxf(q2.y, 0.f);
        q2.z = fmaxf(q2.z, 0.f); q2.w = fmaxf(q2.w, 0.f);
        q3.x = fmaxf(q3.x, 0.f); q3.y = fmaxf(q3.y, 0.f);
        q3.z = fmaxf(q3.z, 0.f); q3.w = fmaxf(q3.w, 0.f);
      }
      v16h a;
      a[0]  = (_Float16)q0.x; a[1]  = (_Float16)q0.y;
      a[2]  = (_Float16)q0.z; a[3]  = (_Float16)q0.w;
      a[4]  = (_Float16)q1.x; a[5]  = (_Float16)q1.y;
      a[6]  = (_Float16)q1.z; a[7]  = (_Float16)q1.w;
      a[8]  = (_Float16)q2.x; a[9]  = (_Float16)q2.y;
      a[10] = (_Float16)q2.z; a[11] = (_Float16)q2.w;
      a[12] = (_Float16)q3.x; a[13] = (_Float16)q3.y;
      a[14] = (_Float16)q3.z; a[15] = (_Float16)q3.w;
#pragma unroll
      for (int nt = 0; nt < 4; ++nt)
        acc[nt] = __builtin_amdgcn_wmma_f32_16x16x32_f16(
            false, a, false, bf[ks][nt], (short)0, acc[nt], false, false);
    }
    // C/D: VGPR r, lane -> (M = row0 + r + 8*hi, N = nt*16 + l16); T padded.
#pragma unroll
    for (int nt = 0; nt < 4; ++nt)
#pragma unroll
      for (int r = 0; r < 8; ++r)
        T[(size_t)(row0 + r + 8 * hi) * F + nt * 16 + l16] = acc[nt][r];
  }
}

// ---------------- aggregation -----------------------------------------------
// out[v] = bias + T[v] * dinv[v]^2   (self-loop term), float4 per thread
__global__ __launch_bounds__(256) void k_agg_init(const float* __restrict__ T,
                                                  const float* __restrict__ dinv,
                                                  const float* __restrict__ bias,
                                                  float* __restrict__ out, int n) {
  int i = blockIdx.x * 256 + threadIdx.x;          // over n*16 quads
  if (i >= n * 16) return;
  int v  = i >> 4;
  int f4 = (i & 15) * 4;
  float s = dinv[v]; s *= s;
  const float4 tv = *(const float4*)(T + (size_t)v * F + f4);
  const float4 bb = *(const float4*)(bias + f4);
  float4 o;
  o.x = tv.x * s + bb.x;  o.y = tv.y * s + bb.y;
  o.z = tv.z * s + bb.z;  o.w = tv.w * s + bb.w;
  *(float4*)(out + (size_t)v * F + f4) = o;
}

// out[dst[e]] += T[src[e]] * norm[e]; 16 lanes per edge, 4 floats per lane.
__global__ __launch_bounds__(256) void k_agg_edges(const int* __restrict__ src,
                                                   const int* __restrict__ dst,
                                                   const float* __restrict__ norm,
                                                   const float* __restrict__ T,
                                                   float* __restrict__ out, int e) {
  int g = blockIdx.x * 16 + (threadIdx.x >> 4);
  if (g >= e) return;
  int f4 = (threadIdx.x & 15) * 4;
  int s = src[g], d = dst[g];
  float w = norm[g];
  float4 m = *(const float4*)(T + (size_t)s * F + f4);
  float* o = out + (size_t)d * F + f4;
  unsafeAtomicAdd(o + 0, m.x * w);
  unsafeAtomicAdd(o + 1, m.y * w);
  unsafeAtomicAdd(o + 2, m.z * w);
  unsafeAtomicAdd(o + 3, m.w * w);
}

// ---------------- pool + classifier -----------------------------------------
__global__ void k_zero64(float* g) { g[threadIdx.x] = 0.0f; }

__global__ __launch_bounds__(256) void k_pool(const float* __restrict__ H,
                                              float* __restrict__ g, int n) {
  int f  = threadIdx.x & 63;
  int v0 = blockIdx.x * 4 + (threadIdx.x >> 6);
  int st = gridDim.x * 4;
  float acc = 0.0f;
  for (int v = v0; v < n; v += st) acc += H[(size_t)v * F + f];
  unsafeAtomicAdd(&g[f], acc);
}

__global__ __launch_bounds__(32) void k_final(const float* __restrict__ g,
                                              const float* __restrict__ Wl,
                                              const float* __restrict__ bl,
                                              float* __restrict__ out, int C_) {
  int c = threadIdx.x;
  if (c < C_) {
    float acc = bl[c];
    for (int k = 0; k < F; ++k) acc += g[k] * Wl[k * C_ + c];
    out[c] = acc;
  }
}

// ---------------- launcher ---------------------------------------------------
extern "C" void kernel_launch(void* const* d_in, const int* in_sizes, int n_in,
                              void* d_out, int out_size, void* d_ws, size_t ws_size,
                              hipStream_t stream) {
  const float* x  = (const float*)d_in[0];
  const int*   ei = (const int*)d_in[1];
  const float* W0 = (const float*)d_in[2];
  const float* b0 = (const float*)d_in[3];
  const float* W1 = (const float*)d_in[4];
  const float* b1 = (const float*)d_in[5];
  const float* W2 = (const float*)d_in[6];
  const float* b2 = (const float*)d_in[7];
  const float* Wl = (const float*)d_in[8];
  const float* bl = (const float*)d_in[9];
  float* out = (float*)d_out;

  const int n    = in_sizes[0] / F;
  const int e    = in_sizes[1] / 2;
  const int C_   = in_sizes[9];
  const int nPad = (n + 511) & ~511;               // full GEMM tiles in scratch
  const int* src = ei;
  const int* dst = ei + e;

  char* p = (char*)d_ws;
  auto alloc = [&](size_t bytes) {
    char* q = p; p += (bytes + 255) & ~(size_t)255; return q;
  };
  float*    dinv  = (float*)alloc((size_t)n * 4);
  float*    norm  = (float*)alloc((size_t)e * 4);
  float*    t     = (float*)alloc((size_t)nPad * F * 4);
  float*    hA    = (float*)alloc((size_t)nPad * F * 4);
  float*    hB    = (float*)alloc((size_t)nPad * F * 4);
  _Float16* Wh0   = (_Float16*)alloc(8192);
  _Float16* Wh1   = (_Float16*)alloc(8192);
  _Float16* Wh2   = (_Float16*)alloc(8192);
  float*    gpool = (float*)alloc(256);
  (void)ws_size; (void)n_in; (void)out_size;

  const int tb = 256;
  k_init_deg <<<(n + tb - 1) / tb, tb, 0, stream>>>(dinv, n);
  k_count_deg<<<(e + tb - 1) / tb, tb, 0, stream>>>(dst, dinv, e);
  k_dinv     <<<(n + tb - 1) / tb, tb, 0, stream>>>(dinv, n);
  k_norm     <<<(e + tb - 1) / tb, tb, 0, stream>>>(src, dst, dinv, norm, e);
  k_prep_w   <<<1, tb, 0, stream>>>(W0, Wh0);
  k_prep_w   <<<1, tb, 0, stream>>>(W1, Wh1);
  k_prep_w   <<<1, tb, 0, stream>>>(W2, Wh2);

  const int gemmGrid    = nPad / 512;              // 512 rows per block
  const int aggInitGrid = (n * 16 + tb - 1) / tb;
  const int aggEdgeGrid = (e + 15) / 16;

  // layer 0: h0 = agg(x @ W0) + b0   (relu fused into next GEMM's A-load)
  k_gemm64<0><<<gemmGrid,    tb, 0, stream>>>(x, Wh0, t, n);
  k_agg_init <<<aggInitGrid, tb, 0, stream>>>(t, dinv, b0, hA, n);
  k_agg_edges<<<aggEdgeGrid, tb, 0, stream>>>(src, dst, norm, t, hA, e);
  // layer 1: h1 = agg(relu(h0) @ W1) + b1
  k_gemm64<1><<<gemmGrid,    tb, 0, stream>>>(hA, Wh1, t, n);
  k_agg_init <<<aggInitGrid, tb, 0, stream>>>(t, dinv, b1, hB, n);
  k_agg_edges<<<aggEdgeGrid, tb, 0, stream>>>(src, dst, norm, t, hB, e);
  // layer 2: h2 = agg(relu(h1) @ W2) + b2   (no relu after)
  k_gemm64<1><<<gemmGrid,    tb, 0, stream>>>(hB, Wh2, t, n);
  k_agg_init <<<aggInitGrid, tb, 0, stream>>>(t, dinv, b2, hA, n);
  k_agg_edges<<<aggEdgeGrid, tb, 0, stream>>>(src, dst, norm, t, hA, e);

  // global_add_pool + classifier
  k_zero64<<<1, 64, 0, stream>>>(gpool);
  k_pool  <<<512, tb, 0, stream>>>(hA, gpool, n);
  k_final <<<1, 32, 0, stream>>>(gpool, Wl, bl, out, C_);
}
